// WeightedSupTConLoss_34522947125365
// MI455X (gfx1250) — matrix operand order repaired and checked by reading, not compiled, not run on hardware
//
#include <hip/hip_runtime.h>
#include <hip/hip_bf16.h>

// WeightedSupTConLoss for MI455X (gfx1250, wave32, WMMA).
//
// Shapes: embeddings [B=256, T=512, C=256] f32, phonemes [B, T] int, weights [K=8] f32.
// Strategy: one 256-thread block per timestep. Normalize rows into an f16 LDS tile
// (131 KB, CDNA5 WGP has 320 KB LDS), then compute the 256x256 Gram matrix with
// v_wmma_f32_16x16x32_f16, exp+mask+row-reduce in registers, deterministic per-class
// aggregation in LDS, per-t loss to workspace, final mean kernel.

typedef __attribute__((ext_vector_type(16))) _Float16 v16h;
typedef __attribute__((ext_vector_type(8)))  float    v8f;

#define B_ 256
#define T_ 512
#define C_ 256
#define K_ 8

// LDS layout (bytes):
//   [0      , 131072) : f16 normalized embeddings [256][256]
//   [131072 , 132096) : int phonemes [256]
//   [132096 , 133120) : float per_sample [256]
//   [133120 , 133152) : float class sums [8]
//   [133152 , 133184) : int class counts [8]
#define SMEM_BYTES 133184

__global__ __launch_bounds__(256, 1)
void suptcon_timestep_kernel(const float* __restrict__ emb,
                             const int*   __restrict__ phon,
                             const float* __restrict__ weights,
                             float*       __restrict__ loss_t_out)
{
    extern __shared__ unsigned char smem_raw[];
    _Float16* sE      = (_Float16*)smem_raw;                // [256][256] f16
    int*      sPh     = (int*)  (smem_raw + 131072);        // [256]
    float*    sPS     = (float*)(smem_raw + 132096);        // [256]
    float*    sClsSum = (float*)(smem_raw + 133120);        // [8]
    int*      sClsCnt = (int*)  (smem_raw + 133152);        // [8]

    const int t    = blockIdx.x;          // timestep
    const int tid  = threadIdx.x;
    const int wave = tid >> 5;            // 8 waves of 32
    const int lane = tid & 31;
    const int lm   = lane & 15;           // lane mod 16
    const int hi   = lane >> 4;           // half-wave selector

    // ---- phonemes for this timestep (one per sample b) ----
    sPh[tid] = phon[(size_t)tid * T_ + t];

    // ---- Phase 1: L2-normalize 256 rows of C=256 f32 -> f16 LDS tile ----
    // One wave per row, 8 floats per lane (two float4 loads, fully coalesced).
    for (int r = wave; r < B_; r += 8) {
        const float* rp = emb + ((size_t)r * T_ + t) * C_ + lane * 8;
        float4 x0 = ((const float4*)rp)[0];
        float4 x1 = ((const float4*)rp)[1];
        float ss = x0.x*x0.x + x0.y*x0.y + x0.z*x0.z + x0.w*x0.w
                 + x1.x*x1.x + x1.y*x1.y + x1.z*x1.z + x1.w*x1.w;
        #pragma unroll
        for (int off = 16; off > 0; off >>= 1) ss += __shfl_xor(ss, off);
        const float scale = 1.0f / fmaxf(sqrtf(ss), 1e-12f);   // F.normalize eps
        _Float16* dp = sE + r * C_ + lane * 8;
        dp[0] = (_Float16)(x0.x * scale);
        dp[1] = (_Float16)(x0.y * scale);
        dp[2] = (_Float16)(x0.z * scale);
        dp[3] = (_Float16)(x0.w * scale);
        dp[4] = (_Float16)(x1.x * scale);
        dp[5] = (_Float16)(x1.y * scale);
        dp[6] = (_Float16)(x1.z * scale);
        dp[7] = (_Float16)(x1.w * scale);
    }
    __syncthreads();

    // ---- Phase 2: Gram matrix via WMMA; exp + mask + row-sums in registers ----
    // Each wave owns two 16-row tiles: rows [rt*16, rt*16+16).
    for (int rt = wave * 2; rt < wave * 2 + 2; ++rt) {
        const int rowBase = rt * 16;

        // Preload the 8 A-fragments (16x32 f16 each) for the full K=256; reused
        // across all 16 column tiles. A-layout: lanes 0-15 hold M=lane with
        // K kc*32+0..15; lanes 16-31 hold K kc*32+16..31 (matches B).
        v16h av[8];
        #pragma unroll
        for (int kc = 0; kc < 8; ++kc) {
            const _Float16* ap = sE + (rowBase + lm) * C_ + kc * 32 + (hi ? 16 : 0);
            #pragma unroll
            for (int i = 0; i < 8; ++i) {
                av[kc][i]     = ap[i];        // VGPR 0..3 : 8 contiguous halves
                av[kc][8 + i] = ap[8 + i];    // VGPR 4..7 : next 8 contiguous halves
            }
        }
        int rowPh[8];
        #pragma unroll
        for (int v = 0; v < 8; ++v) rowPh[v] = sPh[rowBase + v + hi * 8];

        float num[8], den[8];
        #pragma unroll
        for (int v = 0; v < 8; ++v) { num[v] = 0.0f; den[v] = 0.0f; }

        for (int jt = 0; jt < 16; ++jt) {      // column tiles (16 cols each)
            v8f acc = {};
            #pragma unroll
            for (int kc = 0; kc < 8; ++kc) {
                // B-fragment (32x16 f16): lane n = column (= Gram col j), lanes
                // 0-15 hold K kc*32+0..15, lanes 16-31 hold K kc*32+16..31 —
                // 16 contiguous halves of row j in LDS.
                v16h bv;
                const _Float16* bp = sE + (jt * 16 + lm) * C_ + kc * 32 + (hi ? 16 : 0);
                #pragma unroll
                for (int i = 0; i < 16; ++i) bv[i] = bp[i];
                acc = __builtin_amdgcn_wmma_f32_16x16x32_f16(
                        /*neg_a=*/false, av[kc], /*neg_b=*/false, bv,
                        /*c_mod=*/(short)0, acc, /*reuse_a=*/false, /*reuse_b=*/false);
            }
            // D layout: lane -> column N = lm, VGPR v -> row M = v + 8*hi.
            const int col   = jt * 16 + lm;
            const int colPh = sPh[col];
            #pragma unroll
            for (int v = 0; v < 8; ++v) {
                const int  row  = rowBase + v + hi * 8;
                const float e   = __expf(acc[v]);
                const bool offd = (row != col);
                den[v] += offd ? e : 0.0f;
                num[v] += (offd && (rowPh[v] == colPh)) ? e : 0.0f;
            }
        }

        // Reduce num/den across the 16 lanes of each half-wave (xor stays in half).
        #pragma unroll
        for (int v = 0; v < 8; ++v) {
            #pragma unroll
            for (int off = 8; off > 0; off >>= 1) {
                num[v] += __shfl_xor(num[v], off);
                den[v] += __shfl_xor(den[v], off);
            }
        }
        if (lm == 0) {   // lanes 0 and 16 hold totals for 8 rows each
            #pragma unroll
            for (int v = 0; v < 8; ++v)
                sPS[rowBase + v + hi * 8] = -__logf(num[v] / (den[v] + 1e-6f));
        }
    }
    __syncthreads();

    // ---- Phase 3: deterministic per-class weighted mean for this timestep ----
    if (tid < K_) {
        float s = 0.0f; int c = 0;
        for (int b = 0; b < B_; ++b)
            if (sPh[b] == tid) { s += sPS[b]; ++c; }
        sClsSum[tid] = s;
        sClsCnt[tid] = c;
    }
    __syncthreads();
    if (tid == 0) {
        float acc = 0.0f; int present = 0;
        #pragma unroll
        for (int k = 0; k < K_; ++k) {
            if (sClsCnt[k] > 0) {
                acc += (sClsSum[k] / (float)sClsCnt[k]) * weights[k];
                ++present;
            }
        }
        loss_t_out[t] = acc / (float)(present > 0 ? present : 1);
    }
}

// Mean over T per-timestep losses -> scalar output.
__global__ __launch_bounds__(256)
void suptcon_finalize_kernel(const float* __restrict__ loss_t, float* __restrict__ out)
{
    __shared__ float s[256];
    float v = 0.0f;
    for (int i = threadIdx.x; i < T_; i += 256) v += loss_t[i];
    s[threadIdx.x] = v;
    __syncthreads();
    #pragma unroll
    for (int st = 128; st > 0; st >>= 1) {
        if (threadIdx.x < st) s[threadIdx.x] += s[threadIdx.x + st];
        __syncthreads();
    }
    if (threadIdx.x == 0) out[0] = s[0] / (float)T_;
}

extern "C" void kernel_launch(void* const* d_in, const int* in_sizes, int n_in,
                              void* d_out, int out_size, void* d_ws, size_t ws_size,
                              hipStream_t stream)
{
    const float* emb     = (const float*)d_in[0];   // [B,T,C] f32
    const int*   phon    = (const int*)  d_in[1];   // [B,T] int
    const float* weights = (const float*)d_in[2];   // [K] f32
    float*       out     = (float*)d_out;           // scalar
    float*       ws_loss = (float*)d_ws;            // [T] per-timestep losses

    // Dynamic LDS > default cap: raise the limit (no-op / idempotent, not a stream op).
    (void)hipFuncSetAttribute((const void*)suptcon_timestep_kernel,
                              hipFuncAttributeMaxDynamicSharedMemorySize, SMEM_BYTES);

    suptcon_timestep_kernel<<<T_, 256, SMEM_BYTES, stream>>>(emb, phon, weights, ws_loss);
    suptcon_finalize_kernel<<<1, 256, 0, stream>>>(ws_loss, out);
}
